// Attention_27376121545159
// MI455X (gfx1250) — matrix operand order repaired and checked by reading, compiled, not run on hardware
//
#include <hip/hip_runtime.h>
#include <hip/hip_bf16.h>

// ---------------------------------------------------------------------------
// Attention layer forward for MI455X (gfx1250), all matmuls on v_wmma bf16.
// Shapes: B=2, S=2048, H=16, Dh=128, Dm=2048.
// ---------------------------------------------------------------------------

typedef __bf16 bf16;
typedef __attribute__((ext_vector_type(16))) __bf16 bf16x16;
typedef __attribute__((ext_vector_type(8)))  __bf16 bf16x8;
typedef __attribute__((ext_vector_type(8)))  float  f32x8;

union BF16x16 { bf16x16 v; bf16x8 h[2]; };

__device__ __forceinline__ bf16 f2bf(float f) {
    unsigned u = __builtin_bit_cast(unsigned, f);
    unsigned r = (u + 0x7FFFu + ((u >> 16) & 1u)) >> 16;   // RNE truncate
    unsigned short s = (unsigned short)r;
    return __builtin_bit_cast(bf16, s);
}

__device__ __forceinline__ f32x8 wmma_bf16(bf16x16 a, bf16x16 b, f32x8 c) {
    return __builtin_amdgcn_wmma_f32_16x16x32_bf16(false, a, false, b,
                                                   (short)0, c, false, false);
}

// A-fragment (16x32, MxK, row-major memory, ld elements between rows).
// Lane l<16: row=l, K = {0..7, 16..23}; lane>=16: row=l-16, K = {8..15, 24..31}.
__device__ __forceinline__ bf16x16 load_a_frag(const bf16* base, int ld, int lane) {
    const int m    = lane & 15;
    const int koff = (lane < 16) ? 0 : 8;
    BF16x16 r;
    r.h[0] = *(const bf16x8*)(base + (size_t)m * ld + koff);
    r.h[1] = *(const bf16x8*)(base + (size_t)m * ld + 16 + koff);
    return r.v;
}

// B-fragment (32x16, KxN) where memory holds mat[n][k] (n-major, ld elements).
// Lane l<16: col=l, K = {0..7, 16..23}; lane>=16: col=l-16, K = {8..15, 24..31}.
__device__ __forceinline__ bf16x16 load_bT_frag(const bf16* base, int ld, int lane) {
    const int n    = lane & 15;
    const int koff = (lane < 16) ? 0 : 8;
    BF16x16 r;
    r.h[0] = *(const bf16x8*)(base + (size_t)n * ld + koff);
    r.h[1] = *(const bf16x8*)(base + (size_t)n * ld + 16 + koff);
    return r.v;
}

// ---------------------------------------------------------------------------
// Conversion / packing kernels
// ---------------------------------------------------------------------------
__global__ void k_f32_to_bf16(const float* __restrict__ src, bf16* __restrict__ dst, int n) {
    int i = blockIdx.x * blockDim.x + threadIdx.x;
    if (i < n) dst[i] = f2bf(src[i]);
}

// Pack W_Q|W_K|W_V [H,Dm,Dh] into WbT[n][k]: n = t*2048 + h*128 + d (6144), k = Dm (2048)
__global__ void k_pack_wqkv(const float* __restrict__ wq, const float* __restrict__ wk,
                            const float* __restrict__ wv, bf16* __restrict__ wbt) {
    int i = blockIdx.x * blockDim.x + threadIdx.x;
    if (i >= 6144 * 2048) return;
    int n = i >> 11, k = i & 2047;
    int t = n >> 11, rem = n & 2047, h = rem >> 7, d = rem & 127;
    const float* w = (t == 0) ? wq : (t == 1) ? wk : wv;
    wbt[i] = f2bf(w[((size_t)h * 2048 + k) * 128 + d]);
}

// Pack W_O [H,Dh,Dm] into WObT[n][k]: n = out col (2048), k = h*128+d (2048)
__global__ void k_pack_wo(const float* __restrict__ wo, bf16* __restrict__ wobt) {
    int i = blockIdx.x * blockDim.x + threadIdx.x;
    if (i >= 2048 * 2048) return;
    int n = i >> 11, k = i & 2047;
    int h = k >> 7, d = k & 127;
    wobt[i] = f2bf(wo[((size_t)h * 128 + d) * 2048 + n]);
}

// ---------------------------------------------------------------------------
// Fused QKV projection: C[4096x6144] = Ab x WbT, +bias, Q pre-scaled by 1/sqrt(Dh)
// Q,K -> [B,H,S,Dh] bf16 ; V -> transposed [B,H,Dh,S] bf16
// Block = 256 thr (8 waves, 2x4), block tile 128x256, wave tile 64x64 (4x4 frags):
// 16 WMMAs per 16 b128 loads per k-step.
// ---------------------------------------------------------------------------
__global__ void __launch_bounds__(256) k_qkv_proj(
        const bf16* __restrict__ Ab, const bf16* __restrict__ WbT,
        const float* __restrict__ bq, const float* __restrict__ bk,
        const float* __restrict__ bv,
        bf16* __restrict__ Qb, bf16* __restrict__ Kb, bf16* __restrict__ VTb) {
    const int lane = threadIdx.x & 31;
    const int w    = threadIdx.x >> 5;
    const int m0   = blockIdx.y * 128 + (w & 1) * 64;
    const int n0   = blockIdx.x * 256 + (w >> 1) * 64;

    f32x8 acc[4][4] = {};
    for (int kk = 0; kk < 2048; kk += 32) {
        bf16x16 af[4], bg[4];
        #pragma unroll
        for (int i = 0; i < 4; ++i)
            af[i] = load_a_frag(Ab + (size_t)(m0 + 16 * i) * 2048 + kk, 2048, lane);
        #pragma unroll
        for (int j = 0; j < 4; ++j)
            bg[j] = load_bT_frag(WbT + (size_t)(n0 + 16 * j) * 2048 + kk, 2048, lane);
        #pragma unroll
        for (int i = 0; i < 4; ++i)
            #pragma unroll
            for (int j = 0; j < 4; ++j)
                acc[i][j] = wmma_bf16(af[i], bg[j], acc[i][j]);
    }

    const int half = lane >> 4, nl = lane & 15;
    const float qscale = 0.088388347648318447f;   // 1/sqrt(128)
    for (int mi = 0; mi < 4; ++mi)
    for (int ni = 0; ni < 4; ++ni) {
        #pragma unroll
        for (int e = 0; e < 8; ++e) {
            int m = m0 + mi * 16 + e + half * 8;
            int n = n0 + ni * 16 + nl;
            int t = n >> 11, rem = n & 2047, h = rem >> 7, d = rem & 127;
            const float* bias = (t == 0) ? bq : (t == 1) ? bk : bv;
            float v = acc[mi][ni][e] + bias[rem];
            int bb = m >> 11, s = m & 2047;
            if (t == 0) {
                v *= qscale;
                Qb[(((size_t)(bb * 16 + h)) * 2048 + s) * 128 + d] = f2bf(v);
            } else if (t == 1) {
                Kb[(((size_t)(bb * 16 + h)) * 2048 + s) * 128 + d] = f2bf(v);
            } else {
                VTb[(((size_t)(bb * 16 + h)) * 128 + d) * 2048 + s] = f2bf(v);
            }
        }
    }
}

// ---------------------------------------------------------------------------
// Causal flash attention. Grid: (S/128, B*H), block 256 = 8 waves.
// Each wave owns 16 query rows; the block cooperatively stages each 32-key
// K tile (8KB) and V^T tile (8KB) into LDS once, then all 8 waves read their
// WMMA fragments from LDS (8x less global K/V traffic).
// ---------------------------------------------------------------------------
__global__ void __launch_bounds__(256) k_attn(
        const bf16* __restrict__ Qb, const bf16* __restrict__ Kb,
        const bf16* __restrict__ VTb, bf16* __restrict__ Zb) {
    __shared__ __align__(16) bf16 Klds[32][136];    // [key][dh], padded rows
    __shared__ __align__(16) bf16 Vlds[128][40];    // [dh][key], padded rows
    __shared__ __align__(16) bf16 Plds[8][16][40];  // per-wave 16x32 P tile

    const int tid  = threadIdx.x;
    const int lane = tid & 31;
    const int w    = tid >> 5;
    const int bh   = blockIdx.y;               // bb*16 + h
    const int bb   = bh >> 4, h = bh & 15;
    const int qr0  = blockIdx.x * 128 + w * 16;

    const bf16* qptr = Qb  + ((size_t)bh * 2048 + qr0) * 128;
    const bf16* kptr = Kb  + (size_t)bh * 2048 * 128;
    const bf16* vptr = VTb + (size_t)bh * 128  * 2048;

    bf16x16 aQ[4];
    #pragma unroll
    for (int i = 0; i < 4; ++i) aQ[i] = load_a_frag(qptr + 32 * i, 128, lane);

    f32x8 acc[8] = {};
    float mrow[8], lrow[8];
    #pragma unroll
    for (int e = 0; e < 8; ++e) { mrow[e] = -3.0e38f; lrow[e] = 0.0f; }

    const int half = lane >> 4, nl = lane & 15;
    // Uniform trip count for the whole block (needed for __syncthreads);
    // fully-masked trailing tiles are numerically inert (p=0, corr=1).
    const int nkt = blockIdx.x * 4 + 4;

    for (int kt32 = 0; kt32 < nkt; ++kt32) {
        const int kt = kt32 << 5;

        // ---- cooperative stage: K[kt..kt+31][0..127], V^T[0..127][kt..kt+31]
        {
            const int key = tid >> 3, dh0 = (tid & 7) * 16;
            const bf16* gk = kptr + (size_t)(kt + key) * 128 + dh0;
            *(bf16x8*)&Klds[key][dh0]     = *(const bf16x8*)(gk);
            *(bf16x8*)&Klds[key][dh0 + 8] = *(const bf16x8*)(gk + 8);
            const int dh = tid >> 1, ko = (tid & 1) * 16;
            const bf16* gv = vptr + (size_t)dh * 2048 + kt + ko;
            *(bf16x8*)&Vlds[dh][ko]     = *(const bf16x8*)(gv);
            *(bf16x8*)&Vlds[dh][ko + 8] = *(const bf16x8*)(gv + 8);
            if (kt32 + 1 < nkt)
                __builtin_prefetch(kptr + (size_t)(kt + 32 + key) * 128 + dh0, 0, 0);
        }
        __syncthreads();

        // ---- S = Q * K^T  (two 16-key column fragments), K from LDS
        f32x8 s0 = {}, s1 = {};
        #pragma unroll
        for (int i = 0; i < 4; ++i) {
            bf16x16 b0 = load_bT_frag(&Klds[0][32 * i],  136, lane);
            bf16x16 b1 = load_bT_frag(&Klds[16][32 * i], 136, lane);
            s0 = wmma_bf16(aQ[i], b0, s0);
            s1 = wmma_bf16(aQ[i], b1, s1);
        }

        // ---- causal mask (frontier and fully-masked tiles)
        if (kt + 31 > qr0) {
            #pragma unroll
            for (int e = 0; e < 8; ++e) {
                int qrow = qr0 + e + half * 8;
                if (kt + nl      > qrow) s0[e] = -100000.0f;
                if (kt + 16 + nl > qrow) s1[e] = -100000.0f;
            }
        }

        // ---- online softmax: row reductions across each 16-lane half
        #pragma unroll
        for (int e = 0; e < 8; ++e) {
            float t = fmaxf(s0[e], s1[e]);
            t = fmaxf(t, __shfl_xor(t, 1));
            t = fmaxf(t, __shfl_xor(t, 2));
            t = fmaxf(t, __shfl_xor(t, 4));
            t = fmaxf(t, __shfl_xor(t, 8));
            float mnew = fmaxf(mrow[e], t);
            float corr = __expf(mrow[e] - mnew);
            float p0   = __expf(s0[e] - mnew);
            float p1   = __expf(s1[e] - mnew);
            float rs = p0 + p1;
            rs += __shfl_xor(rs, 1);
            rs += __shfl_xor(rs, 2);
            rs += __shfl_xor(rs, 4);
            rs += __shfl_xor(rs, 8);
            lrow[e] = lrow[e] * corr + rs;
            mrow[e] = mnew;
            #pragma unroll
            for (int f = 0; f < 8; ++f) acc[f][e] *= corr;
            int m = e + half * 8;                     // C-layout row
            Plds[w][m][nl]      = f2bf(p0);
            Plds[w][m][16 + nl] = f2bf(p1);
        }
        __builtin_amdgcn_wave_barrier();

        // ---- relayout P (C-frag -> A-frag) via per-wave LDS, O += P * V
        {
            const int m    = nl;
            const int koff = half ? 8 : 0;
            BF16x16 r;
            r.h[0] = *(const bf16x8*)&Plds[w][m][koff];
            r.h[1] = *(const bf16x8*)&Plds[w][m][16 + koff];
            bf16x16 aP = r.v;
            #pragma unroll
            for (int f = 0; f < 8; ++f) {
                bf16x16 bV = load_bT_frag(&Vlds[f * 16][0], 40, lane);
                acc[f] = wmma_bf16(aP, bV, acc[f]);
            }
        }
        __syncthreads();   // protect Klds/Vlds before next stage
    }

    // normalize and store Z as bf16 [B,S,H*Dh]
    #pragma unroll
    for (int e = 0; e < 8; ++e) {
        float inv = 1.0f / lrow[e];
        int s = qr0 + e + half * 8;
        size_t rowbase = ((size_t)bb * 2048 + s) * 2048 + h * 128;
        #pragma unroll
        for (int f = 0; f < 8; ++f)
            Zb[rowbase + f * 16 + nl] = f2bf(acc[f][e] * inv);
    }
}

// ---------------------------------------------------------------------------
// Output projection: out[4096x2048] = Zb x WObT + b_O (f32 result)
// Same 64x64 wave tiling as the QKV projection.
// ---------------------------------------------------------------------------
__global__ void __launch_bounds__(256) k_out_proj(
        const bf16* __restrict__ Zb, const bf16* __restrict__ WObT,
        const float* __restrict__ bO, float* __restrict__ out) {
    const int lane = threadIdx.x & 31;
    const int w    = threadIdx.x >> 5;
    const int m0   = blockIdx.y * 128 + (w & 1) * 64;
    const int n0   = blockIdx.x * 256 + (w >> 1) * 64;

    f32x8 acc[4][4] = {};
    for (int kk = 0; kk < 2048; kk += 32) {
        bf16x16 af[4], bg[4];
        #pragma unroll
        for (int i = 0; i < 4; ++i)
            af[i] = load_a_frag(Zb + (size_t)(m0 + 16 * i) * 2048 + kk, 2048, lane);
        #pragma unroll
        for (int j = 0; j < 4; ++j)
            bg[j] = load_bT_frag(WObT + (size_t)(n0 + 16 * j) * 2048 + kk, 2048, lane);
        #pragma unroll
        for (int i = 0; i < 4; ++i)
            #pragma unroll
            for (int j = 0; j < 4; ++j)
                acc[i][j] = wmma_bf16(af[i], bg[j], acc[i][j]);
    }

    const int half = lane >> 4, nl = lane & 15;
    for (int mi = 0; mi < 4; ++mi)
    for (int ni = 0; ni < 4; ++ni) {
        #pragma unroll
        for (int e = 0; e < 8; ++e) {
            int m = m0 + mi * 16 + e + half * 8;
            int n = n0 + ni * 16 + nl;
            out[(size_t)m * 2048 + n] = acc[mi][ni][e] + bO[n];
        }
    }
}

// ---------------------------------------------------------------------------
extern "C" void kernel_launch(void* const* d_in, const int* in_sizes, int n_in,
                              void* d_out, int out_size, void* d_ws, size_t ws_size,
                              hipStream_t stream) {
    const float* residual = (const float*)d_in[0];
    // d_in[1] ("x") is unused: reference reads QKV from residual.
    const float* wq = (const float*)d_in[2];
    const float* wk = (const float*)d_in[3];
    const float* wv = (const float*)d_in[4];
    const float* wo = (const float*)d_in[5];
    const float* bq = (const float*)d_in[6];
    const float* bk = (const float*)d_in[7];
    const float* bv = (const float*)d_in[8];
    const float* bo = (const float*)d_in[9];
    float* out = (float*)d_out;

    char* ws = (char*)d_ws;
    size_t off = 0;
    auto take = [&](size_t bytes) -> char* {
        char* p = ws + off;
        off += (bytes + 255) & ~(size_t)255;
        return p;
    };
    bf16* Ab   = (bf16*)take((size_t)4096 * 2048 * 2);           // residual bf16
    bf16* WbT  = (bf16*)take((size_t)6144 * 2048 * 2);           // packed QKV weights (n-major)
    bf16* WObT = (bf16*)take((size_t)2048 * 2048 * 2);           // packed O weights (n-major)
    bf16* Qb   = (bf16*)take((size_t)2 * 16 * 2048 * 128 * 2);   // [B,H,S,Dh]
    bf16* Kb   = (bf16*)take((size_t)2 * 16 * 2048 * 128 * 2);   // [B,H,S,Dh]
    bf16* VTb  = (bf16*)take((size_t)2 * 16 * 128 * 2048 * 2);   // [B,H,Dh,S]
    bf16* Zb   = (bf16*)take((size_t)4096 * 2048 * 2);           // [B,S,H*Dh]

    {   // residual -> bf16
        int n = 4096 * 2048;
        k_f32_to_bf16<<<(n + 255) / 256, 256, 0, stream>>>(residual, Ab, n);
    }
    k_pack_wqkv<<<(6144 * 2048) / 256, 256, 0, stream>>>(wq, wk, wv, WbT);
    k_pack_wo  <<<(2048 * 2048) / 256, 256, 0, stream>>>(wo, WObT);

    k_qkv_proj<<<dim3(24, 32), 256, 0, stream>>>(Ab, WbT, bq, bk, bv, Qb, Kb, VTb);
    k_attn    <<<dim3(16, 32), 256, 0, stream>>>(Qb, Kb, VTb, Zb);
    k_out_proj<<<dim3(8, 32),  256, 0, stream>>>(Zb, WObT, bo, out + (size_t)4096 * 2048);

    // first output of the tuple: residual passthrough
    hipMemcpyAsync(d_out, (const void*)residual, (size_t)4096 * 2048 * sizeof(float),
                   hipMemcpyDeviceToDevice, stream);
}
